// Classifier_66116726554918
// MI455X (gfx1250) — matrix operand order repaired and checked by reading, not compiled, not run on hardware
//
#include <hip/hip_runtime.h>
#include <math.h>

typedef __attribute__((ext_vector_type(16))) __bf16 v16bf;
typedef __attribute__((ext_vector_type(8)))  __bf16 v8bf;
typedef __attribute__((ext_vector_type(8)))  float  v8f;

#define BB   16
#define SS   512
#define HH   768
#define NHH  12
#define DHH  64
#define LL   6
#define FFF  3072
#define LABB 10
#define MM   (BB * SS)   // 8192 token rows

// ---------------- fragment helpers ----------------

__device__ __forceinline__ v8f zero8() {
  v8f z;
#pragma unroll
  for (int i = 0; i < 8; ++i) z[i] = 0.0f;
  return z;
}

// 16 consecutive bf16 -> fragment (two b128 loads).
__device__ __forceinline__ v16bf frag16bf(const __bf16* p) {
  const v8bf* q = (const v8bf*)p;
  v8bf lo = q[0], hi = q[1];
  return __builtin_shufflevector(lo, hi, 0, 1, 2, 3, 4, 5, 6, 7, 8, 9, 10, 11, 12, 13, 14, 15);
}

// A-fragment (16x32 bf16) from bf16 memory: p points at (row, k0 + kb).
// elems 0-15 = K=kb..kb+15 (p[0..15]).
__device__ __forceinline__ v16bf frag_a_bf(const __bf16* p) {
  const v8bf* q = (const v8bf*)p;
  v8bf lo = q[0], hi = q[1];
  return __builtin_shufflevector(lo, hi, 0, 1, 2, 3, 4, 5, 6, 7, 8, 9, 10, 11, 12, 13, 14, 15);
}

__device__ __forceinline__ v8f wmma_bf16(v16bf a, v16bf b, v8f c) {
  return __builtin_amdgcn_wmma_f32_16x16x32_bf16(false, a, false, b, (short)0, c, false, false);
}

__device__ __forceinline__ float gelu_tanh(float u) {
  return 0.5f * u * (1.0f + tanhf(0.7978845608028654f * (u + 0.044715f * u * u * u)));
}

// ---------------- weight transpose + bf16 convert: W[K][N] -> Wt[N][K] ----------------

__global__ void __launch_bounds__(256) wtrans_kernel(
    const float* __restrict__ W, __bf16* __restrict__ Wt, int K, int N) {
  int i = blockIdx.x * 256 + threadIdx.x;
  if (i >= K * N) return;
  int n = i / K;
  int k = i - n * K;
  Wt[i] = (__bf16)W[(size_t)k * N + n];
}

// ---------------- V transpose (bf16): V[b,s,h*64+d] -> Vt[(b*NH+h)*64+d][s] ----------------

__global__ void __launch_bounds__(256) vtrans_kernel(
    const __bf16* __restrict__ V, __bf16* __restrict__ Vt) {
  int i = blockIdx.x * 256 + threadIdx.x;   // total = B*NH*DH*S = 6.29M
  int s = i & (SS - 1);
  int rest = i >> 9;
  int d = rest & (DHH - 1);
  int bh = rest >> 6;
  int b = bh / NHH, h = bh - b * NHH;
  Vt[i] = V[((size_t)(b * SS + s)) * HH + h * DHH + d];
}

// ---------------- GEMM: C[M,N] = act(A[M,K] @ W[K,N] + bias) ----------------
// A bf16 (row-major), W pre-transposed bf16 [N][K]; output fp32 or bf16 (outbf).
// 128 threads = 4 waves; wave tile 32(M) x 64(N) = 8 WMMAs per 32-deep K step.
// No LDS, no barriers; all fragments are contiguous b128 pairs.

__global__ void __launch_bounds__(128) gemm_wmma_kernel(
    const __bf16* __restrict__ A, int lda,
    const __bf16* __restrict__ Wt,
    const float* __restrict__ bias,
    float* __restrict__ Cf, __bf16* __restrict__ Cb,
    int Mr, int N, int K, int act, int outbf) {
  const int tid  = threadIdx.x;
  const int wave = tid >> 5;
  const int lane = tid & 31;
  const int rowBase = blockIdx.y * 128 + wave * 32;
  const int colBase = blockIdx.x * 64;
  const int lrow = lane & 15;
  const int kb   = (lane < 16) ? 0 : 16;   // A-fragment K sub-offset
  const int kb2  = (lane < 16) ? 0 : 16;   // B-fragment K sub-offset

  const int lastRow = Mr - 1;
  const int ar0 = min(rowBase + lrow, lastRow);
  const int ar1 = min(rowBase + 16 + lrow, lastRow);
  const __bf16* pa0 = A + (size_t)ar0 * lda + kb;
  const __bf16* pa1 = A + (size_t)ar1 * lda + kb;
  const __bf16* pb0 = Wt + (size_t)(colBase + 0 * 16 + lrow) * K + kb2;
  const __bf16* pb1 = Wt + (size_t)(colBase + 1 * 16 + lrow) * K + kb2;
  const __bf16* pb2 = Wt + (size_t)(colBase + 2 * 16 + lrow) * K + kb2;
  const __bf16* pb3 = Wt + (size_t)(colBase + 3 * 16 + lrow) * K + kb2;

  v8f acc[8];
#pragma unroll
  for (int t = 0; t < 8; ++t) acc[t] = zero8();

#pragma unroll 2
  for (int k0 = 0; k0 < K; k0 += 32) {
    // prefetch two K-steps ahead
    __builtin_prefetch(pa0 + k0 + 64, 0, 0);
    __builtin_prefetch(pb0 + k0 + 64, 0, 0);
    __builtin_prefetch(pb2 + k0 + 64, 0, 0);

    v16bf a0 = frag_a_bf(pa0 + k0);
    v16bf a1 = frag_a_bf(pa1 + k0);
    v16bf b0 = frag16bf(pb0 + k0);
    v16bf b1 = frag16bf(pb1 + k0);
    v16bf b2 = frag16bf(pb2 + k0);
    v16bf b3 = frag16bf(pb3 + k0);
    acc[0] = wmma_bf16(a0, b0, acc[0]);
    acc[4] = wmma_bf16(a1, b0, acc[4]);
    acc[1] = wmma_bf16(a0, b1, acc[1]);
    acc[5] = wmma_bf16(a1, b1, acc[5]);
    acc[2] = wmma_bf16(a0, b2, acc[2]);
    acc[6] = wmma_bf16(a1, b2, acc[6]);
    acc[3] = wmma_bf16(a0, b3, acc[3]);
    acc[7] = wmma_bf16(a1, b3, acc[7]);
  }

  const int rOff = (lane < 16) ? 0 : 8;
#pragma unroll
  for (int g = 0; g < 2; ++g) {
    int rb = rowBase + g * 16;
    if (rb < Mr) {   // Mr is a multiple of 16
#pragma unroll
      for (int t = 0; t < 4; ++t) {
        int col = colBase + t * 16 + lrow;
        float bv = bias[col];
#pragma unroll
        for (int r = 0; r < 8; ++r) {
          int row = rb + r + rOff;
          float v = acc[g * 4 + t][r] + bv;
          if (act == 1)      v = gelu_tanh(v);
          else if (act == 2) v = tanhf(v);
          if (outbf) Cb[(size_t)row * N + col] = (__bf16)v;
          else       Cf[(size_t)row * N + col] = v;
        }
      }
    }
  }
}

// ---------------- Flash attention: one wave per (qtile, head, batch), all-bf16 operands ----------------

__global__ void __launch_bounds__(32) attn_kernel(
    const __bf16* __restrict__ Q, const __bf16* __restrict__ Kb,
    const __bf16* __restrict__ Vt,   // [b*NH+h][d][s]
    const int* __restrict__ seg,
    __bf16* __restrict__ O) {
  __shared__ __bf16 P[16 * 40];  // prob tile 16(q) x 32(k), padded stride
  const int lane = threadIdx.x;
  const int qt = blockIdx.x, h = blockIdx.y, b = blockIdx.z;
  const int lrow = lane & 15;
  const int kb   = (lane < 16) ? 0 : 16;
  const int kb2  = (lane < 16) ? 0 : 16;
  const int rOff = (lane < 16) ? 0 : 8;
  const size_t headOff = (size_t)h * DHH;

  const __bf16* qrow = Q + ((size_t)(b * SS + qt * 16 + lrow)) * HH + headOff;
  v16bf qf0 = frag_a_bf(qrow + kb);        // d = 0..31
  v16bf qf1 = frag_a_bf(qrow + 32 + kb);   // d = 32..63

  float m_i[8], l_i[8];
#pragma unroll
  for (int r = 0; r < 8; ++r) { m_i[r] = -1e30f; l_i[r] = 0.0f; }
  v8f accO[4];
#pragma unroll
  for (int t = 0; t < 4; ++t) accO[t] = zero8();

  const __bf16* vbase = Vt + ((size_t)(b * NHH + h) * DHH) * SS;

  for (int kc = 0; kc < SS; kc += 32) {
    int kcol0 = kc + lrow;
    int kcol1 = kcol0 + 16;
    float mb0 = (seg[b * SS + kcol0] > 0) ? 0.0f : -1e9f;
    float mb1 = (seg[b * SS + kcol1] > 0) ? 0.0f : -1e9f;

    // scores: two 16x16 tiles, K-dim = 64 head dim (2 WMMAs each)
    const __bf16* kp0 = Kb + ((size_t)(b * SS + kcol0)) * HH + headOff + kb2;
    const __bf16* kp1 = Kb + ((size_t)(b * SS + kcol1)) * HH + headOff + kb2;
    v8f s0 = zero8(), s1 = zero8();
    s0 = wmma_bf16(qf0, frag16bf(kp0), s0);
    s0 = wmma_bf16(qf1, frag16bf(kp0 + 32), s0);
    s1 = wmma_bf16(qf0, frag16bf(kp1), s1);
    s1 = wmma_bf16(qf1, frag16bf(kp1 + 32), s1);

    float cr[8];
#pragma unroll
    for (int r = 0; r < 8; ++r) {
      float v0 = s0[r] * 0.125f + mb0;   // 1/sqrt(DH) applied in f32
      float v1 = s1[r] * 0.125f + mb1;
      float mx = fmaxf(v0, v1);
#pragma unroll
      for (int msk = 1; msk < 16; msk <<= 1) mx = fmaxf(mx, __shfl_xor(mx, msk, 32));
      float mnew = fmaxf(m_i[r], mx);
      float c  = __expf(m_i[r] - mnew);
      float p0 = __expf(v0 - mnew);
      float p1 = __expf(v1 - mnew);
      float rs = p0 + p1;
#pragma unroll
      for (int msk = 1; msk < 16; msk <<= 1) rs += __shfl_xor(rs, msk, 32);
      l_i[r] = l_i[r] * c + rs;
      m_i[r] = mnew;
      cr[r]  = c;
      int qrw = r + rOff;
      P[qrw * 40 + lrow]      = (__bf16)p0;
      P[qrw * 40 + lrow + 16] = (__bf16)p1;
    }
#pragma unroll
    for (int t = 0; t < 4; ++t)
#pragma unroll
      for (int r = 0; r < 8; ++r) accO[t][r] *= cr[r];

    __syncthreads();

    // P as A-fragment
    const v8bf* pp = (const v8bf*)&P[lrow * 40 + kb];
    v16bf pa = __builtin_shufflevector(pp[0], pp[1],
        0, 1, 2, 3, 4, 5, 6, 7, 8, 9, 10, 11, 12, 13, 14, 15);

#pragma unroll
    for (int t = 0; t < 4; ++t) {
      const __bf16* vp = vbase + (size_t)(t * 16 + lrow) * SS + kc + kb2;
      accO[t] = wmma_bf16(pa, frag16bf(vp), accO[t]);
    }
    __syncthreads();
  }

#pragma unroll
  for (int t = 0; t < 4; ++t)
#pragma unroll
    for (int r = 0; r < 8; ++r) {
      int qrw = qt * 16 + r + rOff;
      O[((size_t)(b * SS + qrw)) * HH + headOff + t * 16 + lrow] = (__bf16)(accO[t][r] / l_i[r]);
    }
}

// ---------------- block reduction helper (256 threads) ----------------

__device__ __forceinline__ float block_sum256(float v, float* red) {
  int tid = threadIdx.x;
  red[tid] = v;
  __syncthreads();
  for (int s = 128; s > 0; s >>= 1) {
    if (tid < s) red[tid] += red[tid + s];
    __syncthreads();
  }
  float r = red[0];
  __syncthreads();
  return r;
}

// ---------------- embedding gather + LN + selection scale (dual fp32/bf16 output) ----------------

__global__ void __launch_bounds__(256) embed_kernel(
    const int* __restrict__ src, const int* __restrict__ seg, const int* __restrict__ tib,
    const float* __restrict__ wemb, const float* __restrict__ pemb, const float* __restrict__ semb,
    const float* __restrict__ g, const float* __restrict__ bt,
    const float* __restrict__ sel, float* __restrict__ x, __bf16* __restrict__ xb) {
  __shared__ float red[256];
  int t = blockIdx.x;
  int s = t & (SS - 1);
  int sid = src[t], gid = seg[t];
  float vals[3];
  float sum = 0.0f;
#pragma unroll
  for (int i = 0; i < 3; ++i) {
    int j = threadIdx.x + i * 256;
    float v = wemb[(size_t)sid * HH + j] + pemb[(size_t)s * HH + j] + semb[(size_t)gid * HH + j];
    vals[i] = v; sum += v;
  }
  float mean = block_sum256(sum, red) * (1.0f / HH);
  float vs = 0.0f;
#pragma unroll
  for (int i = 0; i < 3; ++i) { float d = vals[i] - mean; vs += d * d; }
  float var = block_sum256(vs, red) * (1.0f / HH);
  float rs = rsqrtf(var + 1e-5f);
  float sf = sel[tib[t]];
#pragma unroll
  for (int i = 0; i < 3; ++i) {
    int j = threadIdx.x + i * 256;
    float v = ((vals[i] - mean) * rs * g[j] + bt[j]) * sf;
    x[(size_t)t * HH + j]  = v;
    xb[(size_t)t * HH + j] = (__bf16)v;
  }
}

// ---------------- residual + LN (in-place into x, dual fp32/bf16 output) ----------------

__global__ void __launch_bounds__(256) resln_kernel(
    float* __restrict__ x, const float* __restrict__ p,
    const float* __restrict__ g, const float* __restrict__ bt,
    __bf16* __restrict__ xb) {
  __shared__ float red[256];
  int t = blockIdx.x;
  float vals[3];
  float sum = 0.0f;
#pragma unroll
  for (int i = 0; i < 3; ++i) {
    int j = threadIdx.x + i * 256;
    float v = x[(size_t)t * HH + j] + p[(size_t)t * HH + j];
    vals[i] = v; sum += v;
  }
  float mean = block_sum256(sum, red) * (1.0f / HH);
  float vs = 0.0f;
#pragma unroll
  for (int i = 0; i < 3; ++i) { float d = vals[i] - mean; vs += d * d; }
  float var = block_sum256(vs, red) * (1.0f / HH);
  float rs = rsqrtf(var + 1e-5f);
#pragma unroll
  for (int i = 0; i < 3; ++i) {
    int j = threadIdx.x + i * 256;
    float v = (vals[i] - mean) * rs * g[j] + bt[j];
    x[(size_t)t * HH + j]  = v;
    xb[(size_t)t * HH + j] = (__bf16)v;
  }
}

// ---------------- logits + log-softmax loss ----------------

__global__ void __launch_bounds__(256) head_kernel(
    const float* __restrict__ hid, const float* __restrict__ Wp2,
    const float* __restrict__ bp2, const int* __restrict__ tgt,
    float* __restrict__ out) {
  __shared__ float lg[BB * LABB];
  int t = threadIdx.x;
  if (t < BB * LABB) {
    int r = t / LABB, c = t - r * LABB;
    float s = bp2[c];
    for (int k = 0; k < HH; ++k) s += hid[r * HH + k] * Wp2[k * LABB + c];
    lg[t] = s;
    out[1 + t] = s;
  }
  __syncthreads();
  if (t == 0) {
    float loss = 0.0f;
    for (int r = 0; r < BB; ++r) {
      float mx = -1e30f;
      for (int c = 0; c < LABB; ++c) mx = fmaxf(mx, lg[r * LABB + c]);
      float se = 0.0f;
      for (int c = 0; c < LABB; ++c) se += __expf(lg[r * LABB + c] - mx);
      float lp = lg[r * LABB + tgt[r]] - mx - logf(se);
      loss -= lp;
    }
    out[0] = loss / BB;
  }
}

// ---------------- host launchers ----------------

static inline void launch_gemm_f(const __bf16* A, int lda, const __bf16* Wt, const float* bias,
                                 float* C, int Mr, int N, int K, int act, hipStream_t st) {
  dim3 g(N / 64, (Mr + 127) / 128);
  gemm_wmma_kernel<<<g, 128, 0, st>>>(A, lda, Wt, bias, C, (__bf16*)nullptr, Mr, N, K, act, 0);
}

static inline void launch_gemm_b(const __bf16* A, int lda, const __bf16* Wt, const float* bias,
                                 __bf16* C, int Mr, int N, int K, int act, hipStream_t st) {
  dim3 g(N / 64, (Mr + 127) / 128);
  gemm_wmma_kernel<<<g, 128, 0, st>>>(A, lda, Wt, bias, (float*)nullptr, C, Mr, N, K, act, 1);
}

static inline void launch_wtrans(const float* W, __bf16* Wt, int K, int N, hipStream_t st) {
  int total = K * N;
  wtrans_kernel<<<(total + 255) / 256, 256, 0, st>>>(W, Wt, K, N);
}

extern "C" void kernel_launch(void* const* d_in, const int* in_sizes, int n_in,
                              void* d_out, int out_size, void* d_ws, size_t ws_size,
                              hipStream_t stream) {
  (void)in_sizes; (void)n_in; (void)out_size; (void)ws_size;
  const int*   src  = (const int*)d_in[0];
  const int*   seg  = (const int*)d_in[1];
  const int*   tgt  = (const int*)d_in[2];
  const int*   tib  = (const int*)d_in[3];
  const float* wemb = (const float*)d_in[4];
  const float* pemb = (const float*)d_in[5];
  const float* semb = (const float*)d_in[6];
  const float* elng = (const float*)d_in[7];
  const float* elnb = (const float*)d_in[8];
  const float* Wq   = (const float*)d_in[9];
  const float* bq   = (const float*)d_in[10];
  const float* Wk   = (const float*)d_in[11];
  const float* bk   = (const float*)d_in[12];
  const float* Wv   = (const float*)d_in[13];
  const float* bv   = (const float*)d_in[14];
  const float* Wo   = (const float*)d_in[15];
  const float* bo   = (const float*)d_in[16];
  const float* l1g  = (const float*)d_in[17];
  const float* l1b  = (const float*)d_in[18];
  const float* Wf1  = (const float*)d_in[19];
  const float* bf1  = (const float*)d_in[20];
  const float* Wf2  = (const float*)d_in[21];
  const float* bf2  = (const float*)d_in[22];
  const float* l2g  = (const float*)d_in[23];
  const float* l2b  = (const float*)d_in[24];
  const float* Wp1  = (const float*)d_in[25];
  const float* bp1  = (const float*)d_in[26];
  const float* Wp2  = (const float*)d_in[27];
  const float* bp2  = (const float*)d_in[28];
  const float* sel  = (const float*)d_in[29];

  float* out = (float*)d_out;
  const size_t MH = (size_t)MM * HH;

  // workspace layout
  float*  x    = (float*)d_ws;             // MH fp32: residual stream
  __bf16* xb   = (__bf16*)(x + MH);        // MH bf16: WMMA copy of x
  __bf16* qbb  = xb + MH;
  __bf16* kbb  = qbb + MH;
  __bf16* vbb  = kbb + MH;
  __bf16* tbb  = vbb + MH;                 // attention output (bf16)
  float*  proj = (float*)(tbb + MH);       // MH fp32: pre-LN projection output
  __bf16* ffb  = (__bf16*)(proj + MH);     // M*FF bf16: gelu output; also Vt scratch in attention
  float*  hid  = (float*)(ffb + (size_t)MM * FFF);
  __bf16* wt   = (__bf16*)(hid + (size_t)BB * HH);   // per-layer bf16 transposed weights

  const size_t oQ = 0, oK = 589824, oV2 = 1179648, oO = 1769472;
  const size_t oF1 = 2359296, oF2 = 4718592;         // layer arena = 7,077,888 halves (~14.2 MB)
  const size_t WH = (size_t)HH * HH, WF = (size_t)HH * FFF;

  embed_kernel<<<MM, 256, 0, stream>>>(src, seg, tib, wemb, pemb, semb, elng, elnb, sel, x, xb);

  for (int l = 0; l < LL; ++l) {
    launch_wtrans(Wq  + l * WH, wt + oQ,  HH,  HH,  stream);
    launch_wtrans(Wk  + l * WH, wt + oK,  HH,  HH,  stream);
    launch_wtrans(Wv  + l * WH, wt + oV2, HH,  HH,  stream);
    launch_wtrans(Wo  + l * WH, wt + oO,  HH,  HH,  stream);
    launch_wtrans(Wf1 + l * WF, wt + oF1, HH,  FFF, stream);
    launch_wtrans(Wf2 + l * WF, wt + oF2, FFF, HH,  stream);

    launch_gemm_b(xb, HH, wt + oQ,  bq + l * HH, qbb, MM, HH, HH, 0, stream);
    launch_gemm_b(xb, HH, wt + oK,  bk + l * HH, kbb, MM, HH, HH, 0, stream);
    launch_gemm_b(xb, HH, wt + oV2, bv + l * HH, vbb, MM, HH, HH, 0, stream);

    vtrans_kernel<<<(int)(MH / 256), 256, 0, stream>>>(vbb, ffb);   // ffb holds Vt
    attn_kernel<<<dim3(SS / 16, NHH, BB), 32, 0, stream>>>(qbb, kbb, ffb, seg, tbb);

    launch_gemm_f(tbb, HH, wt + oO, bo + l * HH, proj, MM, HH, HH, 0, stream);
    resln_kernel<<<MM, 256, 0, stream>>>(x, proj, l1g + l * HH, l1b + l * HH, xb);

    launch_gemm_b(xb, HH, wt + oF1, bf1 + l * FFF, ffb, MM, FFF, HH, 1, stream);
    launch_gemm_f(ffb, FFF, wt + oF2, bf2 + l * HH, proj, MM, HH, FFF, 0, stream);
    resln_kernel<<<MM, 256, 0, stream>>>(x, proj, l2g + l * HH, l2b + l * HH, xb);
  }

  // pooler: rows are xb[:, 0, :] -> lda = S*H gathers them; tanh activation
  launch_wtrans(Wp1, wt + oQ, HH, HH, stream);
  launch_gemm_f(xb, SS * HH, wt + oQ, bp1, hid, BB, HH, HH, 2, stream);
  head_kernel<<<1, 256, 0, stream>>>(hid, Wp2, bp2, tgt, out);
}
